// LightGCN_798863917522
// MI455X (gfx1250) — matrix operand order repaired and checked by reading, not compiled, not run on hardware
//
#include <hip/hip_runtime.h>
#include <hip/hip_bf16.h>

// ---------------------------------------------------------------------------
// LightGCN on MI455X (gfx1250, wave32).
//   - SpMM layers: fp32 atomic scatter (L2-resident tables, 192MB L2).
//   - Scoring GEMM: v_wmma_f32_16x16x32_bf16, one 16x16 tile per wave, K=64
//     (two WMMAs). Output-write bound (410MB fp32) -> WMMA time is hidden.
//   - Sigmoid epilogue uses v_rcp_f32 (1 ULP) instead of IEEE divide chain.
// ---------------------------------------------------------------------------

typedef __bf16 bf16_t;
typedef __attribute__((ext_vector_type(16))) __bf16 v16bf;
typedef __attribute__((ext_vector_type(8)))  __bf16 v8bf;
typedef __attribute__((ext_vector_type(8)))  float  v8f;

#define N_USER  100000
#define M_ITEM  50000
#define DIM     64
#define NNZ     1000000
#define BATCH   2048
#define N_NODES 150000
#define DROPOUT 0.1f

// -------------------------- kernel 1: edge dropout -------------------------
__global__ __launch_bounds__(256) void lgcn_prep_vals(
    const float* __restrict__ edge_val,
    const float* __restrict__ drop_rand,
    float* __restrict__ vals)
{
  int i = blockIdx.x * 256 + threadIdx.x;
  if (i < NNZ) {
    float v = edge_val[i] * (1.0f / (1.0f - DROPOUT));
    vals[i] = (drop_rand[i] >= DROPOUT) ? v : 0.0f;
  }
}

// ------------------- kernel 2: concat embeddings, init acc -----------------
__global__ __launch_bounds__(256) void lgcn_init_emb(
    const float* __restrict__ user_emb,
    const float* __restrict__ item_emb,
    float* __restrict__ e0,
    float* __restrict__ acc)
{
  int i = blockIdx.x * 256 + threadIdx.x;          // over N_NODES*DIM
  if (i < N_NODES * DIM) {
    float v = (i < N_USER * DIM) ? user_emb[i] : item_emb[i - N_USER * DIM];
    e0[i]  = v;
    acc[i] = v;
  }
}

// --------------------- kernel 3: SpMM scatter (one layer) ------------------
// thread = (edge, dim); 64 consecutive threads share one edge -> coalesced
// gather of emb[col] row and coalesced atomic segment on the dest row.
__global__ __launch_bounds__(256) void lgcn_spmm_scatter(
    const float* __restrict__ cur,
    float* __restrict__ nxt,
    const float* __restrict__ vals,
    const int* __restrict__ erow,
    const int* __restrict__ ecol)
{
  long idx = (long)blockIdx.x * 256 + threadIdx.x;  // < NNZ*64
  int e = (int)(idx >> 6);
  int d = (int)(idx & 63);
  float v = vals[e];
  if (v != 0.0f) {
    int r = erow[e];
    int c = ecol[e];
    atomicAdd(&nxt[(long)r * DIM + d], v * cur[(long)c * DIM + d]);
  }
}

// ------------- kernel 4: acc += layer output, zero ping-pong buf -----------
__global__ __launch_bounds__(256) void lgcn_acc_and_zero(
    float* __restrict__ acc,
    const float* __restrict__ src,
    float* __restrict__ zbuf)      // may be null on last layer
{
  int i = blockIdx.x * 256 + threadIdx.x;
  if (i < N_NODES * DIM) {
    acc[i] += src[i];
    if (zbuf) zbuf[i] = 0.0f;
  }
}

// ------- kernel 5a: gather batch users, *0.25 (layer mean), -> bf16 --------
__global__ __launch_bounds__(256) void lgcn_gather_users(
    const float* __restrict__ acc,
    const int* __restrict__ instances,
    bf16_t* __restrict__ u16)
{
  int i = blockIdx.x * 256 + threadIdx.x;          // < BATCH*DIM
  if (i < BATCH * DIM) {
    int b = i >> 6, d = i & 63;
    float v = acc[(long)instances[b] * DIM + d] * 0.25f;
    u16[i] = (bf16_t)v;
  }
}

// ------------- kernel 5b: items *0.25 -> bf16 (contiguous copy) ------------
__global__ __launch_bounds__(256) void lgcn_conv_items(
    const float* __restrict__ acc,
    bf16_t* __restrict__ i16)
{
  int i = blockIdx.x * 256 + threadIdx.x;          // < M_ITEM*DIM
  if (i < M_ITEM * DIM) {
    i16[i] = (bf16_t)(acc[(long)N_USER * DIM + i] * 0.25f);
  }
}

// --------------- kernel 6: WMMA bf16 GEMM + sigmoid epilogue ---------------
// One 16x16 output tile per wave. K = 64 = two v_wmma_f32_16x16x32_bf16.
// A fragment (16x32 bf16, ISA 7.12.2): lane m=lane&15, half=lane>>4:
//   a[i]   = A[m][kb + half*8 + i]       (16B contiguous)
//   a[8+i] = A[m][kb + 16 + half*8 + i]  (16B contiguous)
// B fragment (32x16): lane n=lane&15, khalf=lane>>4, B[k][n]=Items[n][k]:
//   b[i]   = Items[n][kb + khalf*8 + i], b[8+i] = Items[n][kb + 16 + khalf*8 + i]
// C/D: row = 8*half + v, col = lane&15.
__global__ __launch_bounds__(256) void lgcn_gemm_sigmoid(
    const bf16_t* __restrict__ U,     // [BATCH, DIM] bf16
    const bf16_t* __restrict__ I,     // [M_ITEM, DIM] bf16
    float* __restrict__ out)          // [BATCH, M_ITEM] fp32
{
  const int wave = threadIdx.x >> 5;
  const int lane = threadIdx.x & 31;
  const long t   = (long)blockIdx.x * 8 + wave;    // tile id, < 128*3125
  const int tm   = (int)(t & 127);                 // 128 m-tiles (2048/16)
  const int tn   = (int)(t >> 7);                  // 3125 n-tiles (50000/16)
  const int half = lane >> 4;
  const int l16  = lane & 15;

  const bf16_t* arow = U + (long)(tm * 16 + l16) * DIM;
  const bf16_t* brow = I + (long)(tn * 16 + l16) * DIM;

  v8f c = {};
#pragma unroll
  for (int ks = 0; ks < 2; ++ks) {
    const int kb = ks * 32;
    v8bf alo = *(const v8bf*)(arow + kb + half * 8);
    v8bf ahi = *(const v8bf*)(arow + kb + 16 + half * 8);
    v16bf a;
#pragma unroll
    for (int i = 0; i < 8; ++i) { a[i] = alo[i]; a[8 + i] = ahi[i]; }
    v8bf blo = *(const v8bf*)(brow + kb + half * 8);
    v8bf bhi = *(const v8bf*)(brow + kb + 16 + half * 8);
    v16bf b;
#pragma unroll
    for (int i = 0; i < 8; ++i) { b[i] = blo[i]; b[8 + i] = bhi[i]; }
    // (neg_a, A, neg_b, B, c_mod, C, reuse_a, reuse_b)
    c = __builtin_amdgcn_wmma_f32_16x16x32_bf16(
        false, a, false, b, (short)0, c, false, false);
  }

  const int  m0  = tm * 16 + 8 * half;
  const long col = (long)tn * 16 + l16;
#pragma unroll
  for (int v = 0; v < 8; ++v) {
    // sigmoid via v_exp_f32 + v_rcp_f32 (1 ULP), avoids IEEE div chain
    float e = __expf(-c[v]);
    out[(long)(m0 + v) * M_ITEM + col] = __builtin_amdgcn_rcpf(1.0f + e);
  }
}

// ---------------------------------------------------------------------------
extern "C" void kernel_launch(void* const* d_in, const int* in_sizes, int n_in,
                              void* d_out, int out_size, void* d_ws, size_t ws_size,
                              hipStream_t stream) {
  (void)in_sizes; (void)n_in; (void)out_size; (void)ws_size;

  const float* user_emb  = (const float*)d_in[0];
  const float* item_emb  = (const float*)d_in[1];
  const int*   edge_row  = (const int*)  d_in[2];
  const int*   edge_col  = (const int*)  d_in[3];
  const float* edge_val  = (const float*)d_in[4];
  const float* drop_rand = (const float*)d_in[5];
  const int*   instances = (const int*)  d_in[6];
  float*       out       = (float*)d_out;

  // workspace carve-up (all offsets 256B-aligned)
  char* ws = (char*)d_ws;
  const size_t SZ_VALS  = (size_t)NNZ * 4;               //   4.0 MB
  const size_t SZ_EMB   = (size_t)N_NODES * DIM * 4;     //  38.4 MB
  const size_t SZ_U16   = (size_t)BATCH * DIM * 2;       //  0.25 MB
  float*  vals = (float*)(ws);
  float*  e0   = (float*)(ws + SZ_VALS);
  float*  e1   = (float*)(ws + SZ_VALS + SZ_EMB);
  float*  acc  = (float*)(ws + SZ_VALS + 2 * SZ_EMB);
  bf16_t* u16  = (bf16_t*)(ws + SZ_VALS + 3 * SZ_EMB);
  bf16_t* i16  = (bf16_t*)(ws + SZ_VALS + 3 * SZ_EMB + SZ_U16);
  // total ~126 MB

  const int NE   = N_NODES * DIM;                         // 9,600,000
  const int BE   = (NE + 255) / 256;                      // 37,500 blocks
  const long SCAT = (long)NNZ * DIM;                      // 64M threads
  const int BS   = (int)((SCAT + 255) / 256);             // 250,000 blocks

  // 1) dropout
  lgcn_prep_vals<<<(NNZ + 255) / 256, 256, 0, stream>>>(edge_val, drop_rand, vals);
  // 2) init: e0 = acc = concat(user, item); e1 = 0
  lgcn_init_emb<<<BE, 256, 0, stream>>>(user_emb, item_emb, e0, acc);
  hipMemsetAsync(e1, 0, SZ_EMB, stream);

  // 3) layer 1: e0 -> e1 ; acc += e1 ; zero e0
  lgcn_spmm_scatter<<<BS, 256, 0, stream>>>(e0, e1, vals, edge_row, edge_col);
  lgcn_acc_and_zero<<<BE, 256, 0, stream>>>(acc, e1, e0);
  //    layer 2: e1 -> e0 ; acc += e0 ; zero e1
  lgcn_spmm_scatter<<<BS, 256, 0, stream>>>(e1, e0, vals, edge_row, edge_col);
  lgcn_acc_and_zero<<<BE, 256, 0, stream>>>(acc, e0, e1);
  //    layer 3: e0 -> e1 ; acc += e1
  lgcn_spmm_scatter<<<BS, 256, 0, stream>>>(e0, e1, vals, edge_row, edge_col);
  lgcn_acc_and_zero<<<BE, 256, 0, stream>>>(acc, e1, nullptr);

  // 4) fold /4 layer-mean into bf16 conversion; gather batch users
  lgcn_gather_users<<<(BATCH * DIM + 255) / 256, 256, 0, stream>>>(acc, instances, u16);
  lgcn_conv_items<<<(M_ITEM * DIM + 255) / 256, 256, 0, stream>>>(acc, i16);

  // 5) WMMA GEMM + sigmoid: 128*3125 = 400,000 tiles, 8 waves/block
  lgcn_gemm_sigmoid<<<400000 / 8, 256, 0, stream>>>(u16, i16, out);
}
